// mamba_1D_model_80255758893609
// MI455X (gfx1250) — matrix operand order repaired and checked by reading, not compiled, not run on hardware
//
#include <hip/hip_runtime.h>
#include <hip/hip_bf16.h>
#include <math.h>

typedef __attribute__((ext_vector_type(16))) _Float16 v16h;
typedef __attribute__((ext_vector_type(8)))  _Float16 v8h;
typedef __attribute__((ext_vector_type(8)))  float    v8f;

#define D_    128
#define DI_   256
#define DS_   16
#define R_    8
#define DEPTH_ 4
#define NBAND_ 200
#define NCLASS_ 16
#define BATCH_ 64
#define HW_   27
#define P_    9
#define L_    100
#define KC_   4
#define T_    101           // sequence length incl. cls token
#define ROWS1 (BATCH_*T_)   // 6464  (forward sequences)
#define NSEQ  (2*BATCH_)    // 128   (fwd + reversed sequences)
#define ROWS2 (NSEQ*T_)     // 12928
#define EDIM  (R_ + 2*DS_)  // 40

__device__ __forceinline__ float sigmoidf_(float x) { return 1.0f / (1.0f + __expf(-x)); }
__device__ __forceinline__ float siluf_(float x)    { return x * sigmoidf_(x); }
__device__ __forceinline__ float softplusf_(float x) {
    if (x > 20.0f)  return x;
    if (x < -20.0f) return __expf(x);
    return logf(1.0f + __expf(x));
}
__device__ __forceinline__ int revmap_(int l) { return (l < L_) ? (L_ - 1 - l) : L_; }
// sinusoidal positional embedding element (pos, channel o)
__device__ __forceinline__ float pos_embed_(int pos, int o) {
    float div = __expf(-(float)(o & ~1) * (logf(10000.0f) / (float)D_));
    float ang = (float)pos * div;
    return (o & 1) ? cosf(ang) : sinf(ang);
}
__device__ __forceinline__ float blockReduceSum128(float v, float* sm, int tid) {
    sm[tid] = v; __syncthreads();
    for (int s = 64; s > 0; s >>= 1) { if (tid < s) sm[tid] += sm[tid + s]; __syncthreads(); }
    float r = sm[0]; __syncthreads();
    return r;
}

// ---------------------------------------------------------------------------
// f32 -> f16 elementwise (used once per launch for the projection weights)
// ---------------------------------------------------------------------------
__global__ __launch_bounds__(256)
void cvt_f16_kernel(const float* __restrict__ src, _Float16* __restrict__ dst, int n) {
    int i = blockIdx.x * blockDim.x + threadIdx.x;
    if (i < n) dst[i] = (_Float16)src[i];
}

// ---------------------------------------------------------------------------
// WMMA GEMM:  C[M,N] = A[M,K] * W[N,K]^T   (f16 operands in memory, f32 acc)
// Fragments load as aligned 128-bit chunks -> no conversion VALU in the loop.
// One wave computes a 16x64 strip (A fragment reused across 4 WMMAs).
// blockDim = (32,4) = 4 waves stacked in M; grid = (N/64, M/64); K % 32 == 0.
// ---------------------------------------------------------------------------
template<int K>
__global__ __launch_bounds__(128)
void wmma_gemm_nt_kernel(const _Float16* __restrict__ A, const _Float16* __restrict__ W,
                         float* __restrict__ C, int N) {
    const int lane = threadIdx.x;                         // 0..31
    const int m0 = (blockIdx.y * 4 + threadIdx.y) << 4;
    const int n0 = blockIdx.x << 6;
    const int row = lane & 15, hi = lane >> 4;

    v8f acc[4] = {};
    const _Float16* arow = A + (size_t)(m0 + row) * K;
#pragma unroll
    for (int k0 = 0; k0 < K; k0 += 32) {
        // A fragment (16x32): lane(row,hi) holds K = k0+hi*8+0..7 and k0+16+hi*8+0..7
        v8h alo = *(const v8h*)(arow + k0 + hi * 8);
        v8h ahi = *(const v8h*)(arow + k0 + 16 + hi * 8);
        v16h a;
#pragma unroll
        for (int i = 0; i < 8; ++i) { a[i] = alo[i]; a[i + 8] = ahi[i]; }
#pragma unroll
        for (int sub = 0; sub < 4; ++sub) {
            // B fragment (32x16 KxN): lane holds K = k0+hi*8+0..7 and k0+16+hi*8+0..7 of column n0+sub*16+row
            const _Float16* wrow = W + (size_t)(n0 + sub * 16 + row) * K + k0 + hi * 8;
            v8h b0 = *(const v8h*)(wrow);
            v8h b1 = *(const v8h*)(wrow + 16);
            v16h b;
#pragma unroll
            for (int i = 0; i < 8; ++i) { b[i] = b0[i]; b[i + 8] = b1[i]; }
            acc[sub] = __builtin_amdgcn_wmma_f32_16x16x32_f16(
                           false, a, false, b, (short)0, acc[sub], false, false);
        }
    }
    // C/D layout: VGPR j -> element (m0 + j + 8*hi, ncol)
#pragma unroll
    for (int sub = 0; sub < 4; ++sub)
#pragma unroll
        for (int j = 0; j < 8; ++j)
            C[(size_t)(m0 + j + 8 * hi) * N + n0 + sub * 16 + row] = acc[sub][j];
}

// ---------------------------------------------------------------------------
// Patch embed + LN + positional embed + cls token.  grid=(BATCH, T), block=128
// ---------------------------------------------------------------------------
__global__ __launch_bounds__(128)
void embed_kernel(const float* __restrict__ x, const float* __restrict__ pe_w,
                  const float* __restrict__ pe_b, const float* __restrict__ pe_ln_w,
                  const float* __restrict__ pe_ln_b, const float* __restrict__ cls_token,
                  float* __restrict__ t) {
    const int b = blockIdx.x, l = blockIdx.y, o = threadIdx.x;
    __shared__ float sm[128];
    float* trow = t + ((size_t)b * T_ + l) * D_;

    if (l == L_) {  // cls token row (no LN)
        trow[o] = cls_token[o] + pos_embed_(L_, o);
        return;
    }
    float acc = pe_b[o];
    const float* pw = pe_w + (size_t)o * (P_ * P_) * 2;
    const float* x0 = x + ((size_t)b * NBAND_ + 2 * l) * HW_ * HW_;
    const float* x1 = x0 + HW_ * HW_;
#pragma unroll 3
    for (int c = 0; c < P_ * P_; ++c) {
        int off = (9 + c / P_) * HW_ + (9 + c % P_);
        acc += x0[off] * pw[2 * c] + x1[off] * pw[2 * c + 1];
    }
    float mean = blockReduceSum128(acc, sm, o) * (1.0f / D_);
    float dv = acc - mean;
    float var = blockReduceSum128(dv * dv, sm, o) * (1.0f / D_);
    float xn = dv * rsqrtf(var + 1e-5f);
    trow[o] = xn * pe_ln_w[o] + pe_ln_b[o] + pos_embed_(l, o);
}

// ---------------------------------------------------------------------------
// LN (eps 1e-6) of t -> xcat in f16 (GEMM operand).  Forward order only;
// reversal handled downstream since in_proj is per-token.  grid=ROWS1, blk=128.
// ---------------------------------------------------------------------------
__global__ __launch_bounds__(128)
void lnfill_kernel(const float* __restrict__ t, const float* __restrict__ w,
                   const float* __restrict__ bia, _Float16* __restrict__ xcat) {
    const int row = blockIdx.x, c = threadIdx.x;
    __shared__ float sm[128];
    float v = t[(size_t)row * D_ + c];
    float mean = blockReduceSum128(v, sm, c) * (1.0f / D_);
    float dv = v - mean;
    float var = blockReduceSum128(dv * dv, sm, c) * (1.0f / D_);
    xcat[(size_t)row * D_ + c] = (_Float16)(dv * rsqrtf(var + 1e-6f) * w[c] + bia[c]);
}

// ---------------------------------------------------------------------------
// Causal depthwise conv (KC=4) + SiLU on the u half of xz.  Direction 1 reads
// xz rows through the time-reversal remap.  1 thread / (dir,b,l,n) element.
// ---------------------------------------------------------------------------
__global__ __launch_bounds__(256)
void conv_silu_kernel(const float* __restrict__ xz, const float* __restrict__ conv_w,
                      const float* __restrict__ conv_b, float* __restrict__ u) {
    const int idx = blockIdx.x * blockDim.x + threadIdx.x;   // over ROWS2*DI
    const int n = idx & (DI_ - 1), rowg = idx >> 8;          // rowg in [0,ROWS2)
    const int bbg = rowg / T_, l = rowg % T_;
    const int dir = (bbg >= BATCH_), b = bbg - dir * BATCH_;
    const float* w = conv_w + n * KC_;
    float acc = conv_b[n];
#pragma unroll
    for (int k = 0; k < KC_; ++k) {
        int ll = l - (KC_ - 1) + k;
        if (ll >= 0) {
            int srow = b * T_ + (dir ? revmap_(ll) : ll);
            acc += w[k] * xz[(size_t)srow * (2 * DI_) + n];
        }
    }
    u[(size_t)rowg * DI_ + n] = siluf_(acc);
}

// ---------------------------------------------------------------------------
// x_proj: dbl[row, e] = sum_n u[row,n] * xp_w[e,n]   (E=40, K=256)
// ---------------------------------------------------------------------------
__global__ __launch_bounds__(64)
void xproj_kernel(const float* __restrict__ u, const float* __restrict__ xp_w,
                  float* __restrict__ dbl) {
    const int row = blockIdx.x, tid = threadIdx.x;
    __shared__ float su[DI_];
#pragma unroll
    for (int i = 0; i < DI_ / 64; ++i) su[tid + 64 * i] = u[(size_t)row * DI_ + tid + 64 * i];
    __syncthreads();
    if (tid < EDIM) {
        const float* wr = xp_w + (size_t)tid * DI_;
        float acc = 0.0f;
#pragma unroll 8
        for (int n = 0; n < DI_; ++n) acc += su[n] * wr[n];
        dbl[(size_t)row * EDIM + tid] = acc;
    }
}

// ---------------------------------------------------------------------------
// Selective scan.  grid = NSEQ blocks (one per direction*batch sequence),
// 256 threads (channel n).  40-float dbl row staged in LDS each step.
// Emits y directly in f16 (consumed only by the out_proj WMMA GEMM).
// ---------------------------------------------------------------------------
__global__ __launch_bounds__(256)
void scan_kernel(const float* __restrict__ xz, const float* __restrict__ u,
                 const float* __restrict__ dbl, const float* __restrict__ dtp_w,
                 const float* __restrict__ dtp_b, const float* __restrict__ A_log,
                 const float* __restrict__ Dp, _Float16* __restrict__ yact) {
    const int bb = blockIdx.x, n = threadIdx.x;
    const int dir = (bb >= BATCH_), b = bb - dir * BATCH_;
    __shared__ float sdbl[EDIM];

    float dtw[R_];
#pragma unroll
    for (int r = 0; r < R_; ++r) dtw[r] = dtp_w[(size_t)n * R_ + r];
    const float dtb = dtp_b[n];
    const float Dv = Dp[n];
    float Ar[DS_], h[DS_];
#pragma unroll
    for (int s = 0; s < DS_; ++s) { Ar[s] = -__expf(A_log[(size_t)n * DS_ + s]); h[s] = 0.0f; }

    for (int l = 0; l < T_; ++l) {
        const size_t row = (size_t)bb * T_ + l;
        if (n < EDIM) sdbl[n] = dbl[row * EDIM + n];
        __syncthreads();

        float draw = dtb;
#pragma unroll
        for (int r = 0; r < R_; ++r) draw += sdbl[r] * dtw[r];
        const float dt = softplusf_(draw);
        const float uu = u[row * DI_ + n];
        const int zrow = b * T_ + (dir ? revmap_(l) : l);
        const float zz = xz[(size_t)zrow * (2 * DI_) + DI_ + n];
        const float du = dt * uu;
        float y = 0.0f;
#pragma unroll
        for (int s = 0; s < DS_; ++s) {
            h[s] = __expf(dt * Ar[s]) * h[s] + du * sdbl[R_ + s];
            y += h[s] * sdbl[R_ + DS_ + s];
        }
        y = (y + uu * Dv) * siluf_(zz);
        yact[row * DI_ + n] = (_Float16)y;
        __syncthreads();
    }
}

// ---------------------------------------------------------------------------
// Combine fwd + reversed-back outputs, residual add:  t += (y1 + x3)/2
// ---------------------------------------------------------------------------
__global__ __launch_bounds__(256)
void combine_kernel(const float* __restrict__ yout, float* __restrict__ t) {
    const int idx = blockIdx.x * blockDim.x + threadIdx.x;   // over ROWS1*D
    const int c = idx & (D_ - 1), row = idx >> 7;
    const int b = row / T_, l = row % T_;
    float fwd = yout[(size_t)row * D_ + c];
    float bwd = yout[((size_t)(BATCH_ + b) * T_ + revmap_(l)) * D_ + c];
    t[(size_t)row * D_ + c] += 0.5f * (fwd + bwd);
}

// ---------------------------------------------------------------------------
// Mean-pool (tokens 0..99), LN (eps 1e-5), classifier head. 1 block per batch.
// ---------------------------------------------------------------------------
__global__ __launch_bounds__(128)
void head_kernel(const float* __restrict__ t, const float* __restrict__ fcn_w,
                 const float* __restrict__ fcn_b, const float* __restrict__ head_w,
                 const float* __restrict__ head_b, float* __restrict__ out) {
    const int b = blockIdx.x, c = threadIdx.x;
    __shared__ float sm[128];
    __shared__ float sp[128];
    float acc = 0.0f;
    for (int l = 0; l < L_; ++l) acc += t[((size_t)b * T_ + l) * D_ + c];
    acc *= (1.0f / L_);
    float mean = blockReduceSum128(acc, sm, c) * (1.0f / D_);
    float dv = acc - mean;
    float var = blockReduceSum128(dv * dv, sm, c) * (1.0f / D_);
    sp[c] = dv * rsqrtf(var + 1e-5f) * fcn_w[c] + fcn_b[c];
    __syncthreads();
    if (c < NCLASS_) {
        float lg = head_b[c];
        const float* hw = head_w + (size_t)c * D_;
#pragma unroll 8
        for (int k = 0; k < D_; ++k) lg += sp[k] * hw[k];
        out[(size_t)b * NCLASS_ + c] = lg;
    }
}

// ---------------------------------------------------------------------------
extern "C" void kernel_launch(void* const* d_in, const int* in_sizes, int n_in,
                              void* d_out, int out_size, void* d_ws, size_t ws_size,
                              hipStream_t stream) {
    const float* x        = (const float*)d_in[0];
    const float* pe_w     = (const float*)d_in[1];
    const float* pe_b     = (const float*)d_in[2];
    const float* pe_ln_w  = (const float*)d_in[3];
    const float* pe_ln_b  = (const float*)d_in[4];
    const float* cls_tok  = (const float*)d_in[5];
    const float* ln1_w    = (const float*)d_in[6];
    const float* ln1_b    = (const float*)d_in[7];
    const float* in_w     = (const float*)d_in[8];
    const float* conv_w   = (const float*)d_in[9];
    const float* conv_b   = (const float*)d_in[10];
    const float* xp_w     = (const float*)d_in[11];
    const float* dtp_w    = (const float*)d_in[12];
    const float* dtp_b    = (const float*)d_in[13];
    const float* A_log    = (const float*)d_in[14];
    const float* Dp       = (const float*)d_in[15];
    const float* out_w    = (const float*)d_in[16];
    const float* fcn_w    = (const float*)d_in[17];
    const float* fcn_b    = (const float*)d_in[18];
    const float* head_w   = (const float*)d_in[19];
    const float* head_b   = (const float*)d_in[20];
    float* out = (float*)d_out;

    // ---- workspace layout: f32 region, then f16 region -------------------
    float* ws = (float*)d_ws;
    float* t_buf = ws;                              // ROWS1 * D
    float* xz    = t_buf + (size_t)ROWS1 * D_;      // ROWS1 * 2*DI   (forward only)
    float* u_buf = xz    + (size_t)ROWS1 * 2*DI_;   // ROWS2 * DI
    float* dbl   = u_buf + (size_t)ROWS2 * DI_;     // ROWS2 * EDIM
    float* o_buf = dbl   + (size_t)ROWS2 * EDIM;    // ROWS2 * D
    _Float16* h16   = (_Float16*)(o_buf + (size_t)ROWS2 * D_);
    _Float16* xcat16 = h16;                                   // ROWS1 * D
    _Float16* y16    = xcat16 + (size_t)ROWS1 * D_;           // ROWS2 * DI
    _Float16* w16in  = y16    + (size_t)ROWS2 * DI_;          // DEPTH * 2*DI * D
    _Float16* w16out = w16in  + (size_t)DEPTH_ * 2*DI_ * D_;  // DEPTH * D * DI

    // 0) one-time f16 conversion of projection weights (all depths)
    {
        int nin  = DEPTH_ * 2 * DI_ * D_;   // 262144
        int nout = DEPTH_ * D_ * DI_;       // 131072
        cvt_f16_kernel<<<(nin  + 255) / 256, 256, 0, stream>>>(in_w,  w16in,  nin);
        cvt_f16_kernel<<<(nout + 255) / 256, 256, 0, stream>>>(out_w, w16out, nout);
    }

    // 1) patch embed + LN + pos embed + cls
    embed_kernel<<<dim3(BATCH_, T_), 128, 0, stream>>>(x, pe_w, pe_b, pe_ln_w,
                                                       pe_ln_b, cls_tok, t_buf);

    // 2) depth loop
    for (int d = 0; d < DEPTH_; ++d) {
        lnfill_kernel<<<ROWS1, 128, 0, stream>>>(t_buf, ln1_w + d * D_,
                                                 ln1_b + d * D_, xcat16);
        // in_proj (per-token, computed once for both directions):
        // (ROWS1 x 128) x (512 x 128)^T -> xz
        wmma_gemm_nt_kernel<D_><<<dim3((2*DI_)/64, ROWS1/64), dim3(32, 4), 0, stream>>>(
            xcat16, w16in + (size_t)d * 2*DI_ * D_, xz, 2*DI_);

        conv_silu_kernel<<<(ROWS2 * DI_) / 256, 256, 0, stream>>>(
            xz, conv_w + (size_t)d * DI_ * KC_, conv_b + (size_t)d * DI_, u_buf);

        xproj_kernel<<<ROWS2, 64, 0, stream>>>(u_buf, xp_w + (size_t)d * EDIM * DI_, dbl);

        scan_kernel<<<NSEQ, DI_, 0, stream>>>(
            xz, u_buf, dbl, dtp_w + (size_t)d * DI_ * R_, dtp_b + (size_t)d * DI_,
            A_log + (size_t)d * DI_ * DS_, Dp + (size_t)d * DI_, y16);

        // out_proj: (ROWS2 x 256) x (128 x 256)^T -> o_buf
        wmma_gemm_nt_kernel<DI_><<<dim3(D_/64, ROWS2/64), dim3(32, 4), 0, stream>>>(
            y16, w16out + (size_t)d * D_ * DI_, o_buf, D_);

        combine_kernel<<<(ROWS1 * D_) / 256, 256, 0, stream>>>(o_buf, t_buf);
    }

    // 3) pool + LN + head
    head_kernel<<<BATCH_, 128, 0, stream>>>(t_buf, fcn_w, fcn_b, head_w, head_b, out);
}
